// UrLstmRNN_65171833749537
// MI455X (gfx1250) — matrix operand order repaired and checked by reading, not compiled, not run on hardware
//
#include <hip/hip_runtime.h>
#include <hip/hip_bf16.h>

// ---------------------------------------------------------------------------
// UR-LSTM on MI455X (gfx1250), compile-only tuning target.
//
// Fused [x_t | h] @ [W_ih | W_hh]^T per step (K=1024), persistent kernel with
// 32 WGs x 256 threads (8 wave32s). WG w owns hidden units [16w, 16w+16); its
// 64 gate-columns of bf16 weights are DMA'd into 128KB LDS once via the
// Tensor Data Mover and reused for all 512 steps. Each wave owns batch tile
// [16*wave,+16) and keeps the fp32 cell state for its 16x16 patch in the WMMA
// accumulator layout across all timesteps. h is double-buffered in global
// (bf16, L2-resident) with one grid barrier per step. xs is pre-converted to
// bf16 once (if workspace allows) so the steady-state inner loop is pure
// global_load_b128 / ds_load_b128 / v_wmma.
// ---------------------------------------------------------------------------

typedef __attribute__((ext_vector_type(16))) __bf16    v16bf;
typedef __attribute__((ext_vector_type(8)))  float     v8f;
typedef __attribute__((ext_vector_type(4)))  unsigned  u32x4;
typedef __attribute__((ext_vector_type(8)))  int       i32x8;
typedef __attribute__((ext_vector_type(4)))  int       i32x4;

#define INPUT_SZ   512
#define HIDDEN_SZ  512
#define OUTPUT_SZ  256
#define BATCH_SZ   128
#define SEQ_LEN    512
#define KTOT       (INPUT_SZ + HIDDEN_SZ)     // 1024
#define GATES      (4 * HIDDEN_SZ)            // 2048
#define NWG        32                         // 512 hidden / 16 per WG
#define HC         16                         // hidden units per WG
#define WG_COLS    (4 * HC)                   // 64 gate columns per WG

static __device__ __forceinline__ float sigmoidf_(float x) {
    return 1.0f / (1.0f + __expf(-x));
}

static __device__ __forceinline__ float tanhf_(float x) {
    return 2.0f / (1.0f + __expf(-2.0f * x)) - 1.0f;
}

// A fragment (16x32 bf16, M x K) from an fp32 row-major source.
// Layout: lanes 0-15 -> M=lane, K runs {k0..k0+7, k0+16..k0+23}, k0 = 8*(lane>>4).
static __device__ __forceinline__ v16bf load_a_f32(const float* __restrict__ src,
                                                   long ld, int kc) {
    const int lane = threadIdx.x & 31;
    const int m    = lane & 15;
    const int k0   = kc + ((lane >> 4) << 3);
    const float* p = src + (long)m * ld + k0;
    v16bf a;
#pragma unroll
    for (int j = 0; j < 8; ++j) a[j]     = (__bf16)p[j];
#pragma unroll
    for (int j = 0; j < 8; ++j) a[8 + j] = (__bf16)p[16 + j];
    return a;
}

// Same, from a bf16 row-major source (pre-converted xs or the h buffer):
// two 16B contiguous runs per lane -> two global_load_b128.
static __device__ __forceinline__ v16bf load_a_bf16(const __bf16* __restrict__ src,
                                                    long ld, int kc) {
    const int lane = threadIdx.x & 31;
    const int m    = lane & 15;
    const int k0   = kc + ((lane >> 4) << 3);
    const __bf16* p = src + (long)m * ld + k0;
    v16bf a;
#pragma unroll
    for (int j = 0; j < 8; ++j) a[j]     = p[j];
#pragma unroll
    for (int j = 0; j < 8; ++j) a[8 + j] = p[16 + j];
    return a;
}

// B fragment (32x16 bf16, K x N) from the LDS weight slice: each lane reads
// K {k0..k0+7, k0+16..k0+23}, k0 = 8*(lane>>4) -> two ds_load_b128 per fragment.
static __device__ __forceinline__ v16bf load_b_lds(const __bf16* __restrict__ wl,
                                                   int ntile, int kc) {
    const int lane = threadIdx.x & 31;
    const int n    = lane & 15;
    const int k0   = kc + ((lane >> 4) << 3);
    const __bf16* p = wl + (ntile * 16 + n) * KTOT + k0;
    v16bf b;
#pragma unroll
    for (int j = 0; j < 8; ++j) b[j]     = p[j];
#pragma unroll
    for (int j = 0; j < 8; ++j) b[8 + j] = p[16 + j];
    return b;
}

static __device__ __forceinline__ v8f wmma_bf16(v16bf a, v16bf b, v8f c) {
    return __builtin_amdgcn_wmma_f32_16x16x32_bf16(
        /*neg_a=*/false, a, /*neg_b=*/false, b,
        /*c_mod=*/(short)0, c, /*reuse_a=*/false, /*reuse_b=*/false);
}

// --------------------------- setup kernels ---------------------------------

__global__ void urlstm_init(__bf16* h0, __bf16* h1, unsigned* cnt) {
    const int i = blockIdx.x * blockDim.x + threadIdx.x;
    if (i == 0) *cnt = 0u;
    if (i < BATCH_SZ * HIDDEN_SZ) {
        h0[i] = (__bf16)0.0f;
        h1[i] = (__bf16)0.0f;
    }
}

// Fuse [W_ih | W_hh] into one bf16 matrix Wc[2048][1024].
__global__ void urlstm_cvt_weights(const float* __restrict__ Wih,
                                   const float* __restrict__ Whh,
                                   __bf16* __restrict__ Wc) {
    const int i = blockIdx.x * blockDim.x + threadIdx.x;
    if (i >= GATES * KTOT) return;
    const int g = i >> 10;
    const int k = i & (KTOT - 1);
    const float v = (k < INPUT_SZ) ? Wih[g * INPUT_SZ + k]
                                   : Whh[g * HIDDEN_SZ + (k - INPUT_SZ)];
    Wc[i] = (__bf16)v;
}

// Pre-convert xs (fp32) to bf16 once, so the recurrent kernel's x-phase has
// no VALU conversion work (33.5M elements, read/written once).
__global__ void urlstm_cvt_x(const float4* __restrict__ xs4,
                             __bf16* __restrict__ xbf) {
    const long i = (long)blockIdx.x * blockDim.x + threadIdx.x;  // float4 index
    const float4 v = xs4[i];
    __bf16* o = xbf + i * 4;
    o[0] = (__bf16)v.x; o[1] = (__bf16)v.y;
    o[2] = (__bf16)v.z; o[3] = (__bf16)v.w;
}

// --------------------------- persistent recurrence -------------------------

__global__ __launch_bounds__(256) void urlstm_persistent(
    const float*  __restrict__ xs,      // [B, T, 512] fp32 (fallback path)
    const __bf16* __restrict__ xbf,     // [B, T, 512] bf16 or nullptr
    const __bf16* __restrict__ Wc,      // [2048, 1024] bf16
    const float*  __restrict__ bias,    // [2048]
    const float*  __restrict__ fbias,   // [512]
    __bf16* __restrict__ h0,            // [B, 512] ping
    __bf16* __restrict__ h1,            // [B, 512] pong
    float*  __restrict__ hT,            // [B, 512] final fp32 h
    unsigned* __restrict__ bar)
{
    extern __shared__ __bf16 wl[];      // 64 cols x 1024 K = 128KB
    const int tid   = threadIdx.x;
    const int wave  = tid >> 5;
    const int lane  = tid & 31;
    const int wg    = blockIdx.x;       // 0..31
    const int hbase = wg * HC;

    // ---- TDM: DMA this WG's weight slice into LDS (once, reused 512 steps).
    // One 2D tensor_load_to_lds per gate: tile = 16 rows x 2048B, row stride
    // 2048B, data_size = 8B units. lds_addr = low 32 bits of the flat shared
    // pointer (LDS aperture puts the byte offset in addr[31:0]).
    if (tid == 0) {
        const unsigned lds_base = (unsigned)(uintptr_t)(void*)wl;
#pragma unroll
        for (int gi = 0; gi < 4; ++gi) {
            const unsigned long long gaddr =
                (unsigned long long)(uintptr_t)(Wc + (size_t)(gi * HIDDEN_SZ + hbase) * KTOT);
            u32x4 g0;
            g0[0] = 1u;                                        // count=1 user D#
            g0[1] = lds_base + (unsigned)gi * (HC * KTOT * 2); // lds_addr
            g0[2] = (unsigned)(gaddr & 0xFFFFFFFFu);           // global_addr lo
            g0[3] = (unsigned)((gaddr >> 32) & 0x1FFFFFFu)     // global_addr hi
                    | (2u << 30);                              // type = 2 (image)
            i32x8 g1;
            g1[0] = (int)(3u << 16);     // data_size = 3 (8-byte units)
            g1[1] = (int)(256u << 16);   // tensor_dim0 = 256 units (2KB row)
            g1[2] = (int)(16u << 16);    // tensor_dim1 = 16 rows
            g1[3] = (int)(256u << 16);   // tile_dim0 = 256 units
            g1[4] = (int)16;             // tile_dim1 = 16 rows, tile_dim2 = 0
            g1[5] = (int)256;            // tensor_dim0_stride = 256 units
            g1[6] = 0;                   // stride hi / dim1 stride lo
            g1[7] = 0;                   // dim1 stride hi
            i32x4 gz4 = {0, 0, 0, 0};    // groups 2/3 unused (2D tensor)
            i32x8 gz8 = {0, 0, 0, 0, 0, 0, 0, 0};
            // 6-arg form (this toolchain): (g0, g1, g2, g3, g_extra, cpol)
            __builtin_amdgcn_tensor_load_to_lds(g0, g1, gz4, gz4, gz8, 0);
        }
        __builtin_amdgcn_s_wait_tensorcnt(0);
    }
    __syncthreads();

    // ---- per-lane gate constants (fixed across steps) ----
    const int   cn = hbase + (lane & 15);
    const float fb = fbias[cn];
    const float bF = bias[0 * HIDDEN_SZ + cn];
    const float bR = bias[1 * HIDDEN_SZ + cn];
    const float bU = bias[2 * HIDDEN_SZ + cn];
    const float bO = bias[3 * HIDDEN_SZ + cn];

    const int  bm  = wave * 16;                  // batch tile
    const long xld = (long)SEQ_LEN * INPUT_SZ;   // row stride in xs / xbf

    v8f c_state = {};                            // fp32 cell state, in regs

    for (int t = 0; t < SEQ_LEN; ++t) {
        const __bf16* hin  = (t & 1) ? h1 : h0;
        __bf16*       hout = (t & 1) ? h0 : h1;

        v8f aF = {}, aR = {}, aU = {}, aO = {};

        // prefetch next timestep's x rows for this batch tile
        if (t + 1 < SEQ_LEN && lane < 16) {
            if (xbf) __builtin_prefetch(xbf + (long)(bm + lane) * xld
                                            + (long)(t + 1) * INPUT_SZ, 0, 1);
            else     __builtin_prefetch(xs  + (long)(bm + lane) * xld
                                            + (long)(t + 1) * INPUT_SZ, 0, 1);
        }

        // K = 0..511 : input contribution from xs
        if (xbf) {   // uniform branch; steady-state path, no conversions
            const __bf16* xsrc = xbf + (long)bm * xld + (long)t * INPUT_SZ;
#pragma unroll 2
            for (int kc = 0; kc < INPUT_SZ; kc += 32) {
                const v16bf a = load_a_bf16(xsrc, xld, kc);
                aF = wmma_bf16(a, load_b_lds(wl, 0, kc), aF);
                aR = wmma_bf16(a, load_b_lds(wl, 1, kc), aR);
                aU = wmma_bf16(a, load_b_lds(wl, 2, kc), aU);
                aO = wmma_bf16(a, load_b_lds(wl, 3, kc), aO);
            }
        } else {     // fallback: convert on the fly
            const float* xsrc = xs + (long)bm * xld + (long)t * INPUT_SZ;
#pragma unroll 2
            for (int kc = 0; kc < INPUT_SZ; kc += 32) {
                const v16bf a = load_a_f32(xsrc, xld, kc);
                aF = wmma_bf16(a, load_b_lds(wl, 0, kc), aF);
                aR = wmma_bf16(a, load_b_lds(wl, 1, kc), aR);
                aU = wmma_bf16(a, load_b_lds(wl, 2, kc), aU);
                aO = wmma_bf16(a, load_b_lds(wl, 3, kc), aO);
            }
        }
        // K = 512..1023 : recurrent contribution from h (bf16)
        const __bf16* hsrc = hin + bm * HIDDEN_SZ;
#pragma unroll 2
        for (int kc = 0; kc < HIDDEN_SZ; kc += 32) {
            const v16bf a = load_a_bf16(hsrc, HIDDEN_SZ, kc);
            aF = wmma_bf16(a, load_b_lds(wl, 0, INPUT_SZ + kc), aF);
            aR = wmma_bf16(a, load_b_lds(wl, 1, INPUT_SZ + kc), aR);
            aU = wmma_bf16(a, load_b_lds(wl, 2, INPUT_SZ + kc), aU);
            aO = wmma_bf16(a, load_b_lds(wl, 3, INPUT_SZ + kc), aO);
        }

        // ---- UR-LSTM gate math (fp32, accumulator layout) ----
        // C layout: lanes 0-15: M=e, N=lane; lanes 16-31: M=e+8, N=lane-16.
#pragma unroll
        for (int e = 0; e < 8; ++e) {
            const float f = sigmoidf_(aF[e] + bF + fb);
            const float r = sigmoidf_(aR[e] + bR - fb);
            const float g = 2.0f * r * f + (1.0f - 2.0f * r) * f * f;
            const float u = tanhf_(aU[e] + bU);
            const float cc = g * c_state[e] + (1.0f - g) * u;
            c_state[e] = cc;
            const float h = sigmoidf_(aO[e] + bO) * tanhf_(cc);
            const int b = bm + e + ((lane >> 4) << 3);
            hout[b * HIDDEN_SZ + cn] = (__bf16)h;
            if (t == SEQ_LEN - 1) hT[b * HIDDEN_SZ + cn] = h;
        }

        // ---- one grid barrier per step (monotonic counter) ----
        __threadfence();
        __syncthreads();
        if (tid == 0) {
            atomicAdd(bar, 1u);
            const unsigned target = (unsigned)(t + 1) * NWG;
            while (__hip_atomic_load(bar, __ATOMIC_ACQUIRE,
                                     __HIP_MEMORY_SCOPE_AGENT) < target) {
                __builtin_amdgcn_s_sleep(1);
            }
        }
        __syncthreads();
    }
}

// --------------------------- output projection -----------------------------

__global__ void urlstm_outproj(const float* __restrict__ hT,
                               const float* __restrict__ Wout,
                               const float* __restrict__ bout,
                               float* __restrict__ out) {
    const int i = blockIdx.x * blockDim.x + threadIdx.x;
    if (i >= BATCH_SZ * OUTPUT_SZ) return;
    const int b = i >> 8;
    const int o = i & (OUTPUT_SZ - 1);
    const float* hp = hT + b * HIDDEN_SZ;
    const float* wp = Wout + o * HIDDEN_SZ;
    float acc = bout[o];
#pragma unroll 4
    for (int k = 0; k < HIDDEN_SZ; ++k) acc = fmaf(hp[k], wp[k], acc);
    out[b * OUTPUT_SZ + o] = acc;
}

// --------------------------- launcher --------------------------------------

extern "C" void kernel_launch(void* const* d_in, const int* in_sizes, int n_in,
                              void* d_out, int out_size, void* d_ws, size_t ws_size,
                              hipStream_t stream) {
    const float* xs    = (const float*)d_in[0];   // [128,512,512]
    const float* W_ih  = (const float*)d_in[1];   // [2048,512]
    const float* W_hh  = (const float*)d_in[2];   // [2048,512]
    const float* bias  = (const float*)d_in[3];   // [2048]
    const float* fbias = (const float*)d_in[4];   // [512]
    const float* W_out = (const float*)d_in[5];   // [256,512]
    const float* b_out = (const float*)d_in[6];   // [256]
    float* out = (float*)d_out;                   // [128,256]

    // workspace layout
    char* ws = (char*)d_ws;
    size_t off = 0;
    __bf16* Wc = (__bf16*)(ws + off); off += (size_t)GATES * KTOT * 2;          // 4MB
    __bf16* h0 = (__bf16*)(ws + off); off += (size_t)BATCH_SZ * HIDDEN_SZ * 2;  // 128KB
    __bf16* h1 = (__bf16*)(ws + off); off += (size_t)BATCH_SZ * HIDDEN_SZ * 2;  // 128KB
    float*  hT = (float*)(ws + off);  off += (size_t)BATCH_SZ * HIDDEN_SZ * 4;  // 256KB
    unsigned* bar = (unsigned*)(ws + off); off += 256;                          // padded
    const size_t xn = (size_t)BATCH_SZ * SEQ_LEN * INPUT_SZ;                    // 33.5M
    __bf16* xbf = nullptr;
    if (ws_size >= off + xn * sizeof(__bf16)) xbf = (__bf16*)(ws + off);        // 67MB

    // 1) zero h double-buffers + barrier counter (ws is poisoned by harness)
    urlstm_init<<<(BATCH_SZ * HIDDEN_SZ + 255) / 256, 256, 0, stream>>>(h0, h1, bar);

    // 2) fuse + convert weights to bf16
    urlstm_cvt_weights<<<(GATES * KTOT + 255) / 256, 256, 0, stream>>>(W_ih, W_hh, Wc);

    // 2b) pre-convert xs to bf16 if workspace allows
    if (xbf) {
        urlstm_cvt_x<<<(unsigned)(xn / 4 / 256), 256, 0, stream>>>(
            (const float4*)xs, xbf);
    }

    // 3) persistent recurrence: 32 WGs x 256 threads, 128KB LDS each
    urlstm_persistent<<<NWG, 256, WG_COLS * KTOT * sizeof(__bf16), stream>>>(
        xs, xbf, Wc, bias, fbias, h0, h1, hT, bar);

    // 4) output projection (tiny, fp32)
    urlstm_outproj<<<(BATCH_SZ * OUTPUT_SZ + 255) / 256, 256, 0, stream>>>(
        hT, W_out, b_out, out);
}
